// StyleGAN2_3D_Layer_70806830842217
// MI455X (gfx1250) — matrix operand order repaired and checked by reading, not compiled, not run on hardware
//
#include <hip/hip_runtime.h>
#include <hip/hip_bf16.h>

typedef __bf16 bf16;
typedef __attribute__((ext_vector_type(16))) __bf16        v16bf;
typedef __attribute__((ext_vector_type(8)))  __bf16        v8bf;
typedef __attribute__((ext_vector_type(8)))  float         v8f;
typedef __attribute__((ext_vector_type(8)))  float         f32x8;
typedef __attribute__((ext_vector_type(8)))  unsigned int  u32x8;

#define CIN   32
#define COUT  32
#define KOFF  27
#define NB    8
#define WDIM  512
#define EPSV  1e-8f
#define SLOPE 0.2f
#define TILES_PER_BLOCK 8

// ---------------------------------------------------------------------------
// 0) zero the stats accumulators in workspace
// ---------------------------------------------------------------------------
__global__ void zero_ws_kernel(float* ws, int nfloats) {
    int i = blockIdx.x * blockDim.x + threadIdx.x;
    if (i < nfloats) ws[i] = 0.0f;
}

// ---------------------------------------------------------------------------
// 0b) one-shot feats fp32 -> bf16 (row-major [N,32], 64B rows)
// ---------------------------------------------------------------------------
__global__ __launch_bounds__(256) void cvt_feats_kernel(
    const float* __restrict__ feats, bf16* __restrict__ fb, int total8)
{
    int i = blockIdx.x * 256 + threadIdx.x;     // one thread = 8 elements
    if (i >= total8) return;
    f32x8 v = *(const f32x8*)(feats + (long long)i * 8);
    v8bf  h = __builtin_convertvector(v, v8bf);
    *(v8bf*)(fb + (long long)i * 8) = h;
}

// ---------------------------------------------------------------------------
// 1) sparse conv via V_WMMA_F32_16X16X32_BF16
//    each wave32 computes a 16-voxel x 32-channel output tile
//    BF16FEATS: gather pre-converted bf16 rows (2x b128/lane/offset)
// ---------------------------------------------------------------------------
template <bool BF16FEATS>
__global__ __launch_bounds__(256) void conv_wmma_kernel(
    const void* __restrict__ featsv,   // [N, 32] f32 or bf16
    const float* __restrict__ convW,   // [27, 32, 32]  (k, cin, cout)
    const float* __restrict__ convB,   // [32]
    const int*   __restrict__ nidx,    // [N, 27]
    float*       __restrict__ xout,    // [N, 32]
    int n, int nTiles)
{
    // LDS weights as packed bf16 pairs, layout [k][cout][cin/2]:
    // word j of row (k,cout) = { cin=2j (lo16), cin=2j+1 (hi16) }  == B VGPR j
    __shared__ unsigned int ldsW[KOFF * COUT * (CIN / 2)];

    const int tid = threadIdx.x;
    for (int i = tid; i < KOFF * COUT * (CIN / 2); i += 256) {
        int k    = i / (COUT * (CIN / 2));
        int r    = i - k * (COUT * (CIN / 2));
        int cout = r >> 4;            // /16
        int cin  = (r & 15) * 2;      // cin pair
        const float* src = convW + ((long long)k * CIN + cin) * COUT + cout;
        bf16 lo = (bf16)src[0];
        bf16 hi = (bf16)src[COUT];
        unsigned int pk = (unsigned int)__builtin_bit_cast(unsigned short, lo) |
                          ((unsigned int)__builtin_bit_cast(unsigned short, hi) << 16);
        ldsW[i] = pk;
    }
    __syncthreads();

    const int wave = tid >> 5;
    const int lane = tid & 31;
    const int half = lane >> 4;   // 0: lanes 0-15, 1: lanes 16-31
    const int lm   = lane & 15;   // A: row M ; B/C/D: column N

    const int tile = blockIdx.x * TILES_PER_BLOCK + wave;
    if (tile >= nTiles) return;   // wave-uniform: EXEC stays all-ones for WMMA
    const int base = tile * 16;

    // accumulators initialized with conv bias (same value in all 8 rows held)
    const float cb0 = convB[lm];
    const float cb1 = convB[16 + lm];
    v8f acc0, acc1;
#pragma unroll
    for (int r = 0; r < 8; ++r) { acc0[r] = cb0; acc1[r] = cb1; }

    int arow = base + lm;                    // voxel row this lane feeds into A
    if (arow > n - 1) arow = n - 1;          // tail clamp (data only, no divergence)
    const int* __restrict__ nrow = nidx + (long long)arow * KOFF;

    // prefetch all 27 neighbor indices (streamed once -> non-temporal)
    int gidx[KOFF];
#pragma unroll
    for (int k = 0; k < KOFF; ++k) gidx[k] = __builtin_nontemporal_load(nrow + k);

    const char* fbase = (const char*)featsv;

#pragma unroll
    for (int k = 0; k < KOFF; ++k) {
        // A fragment (contiguous k order, matched by B):
        //   lanes 0-15:  K = 0..15
        //   lanes 16-31: K = 16..31
        v16bf a;
        if (BF16FEATS) {
            // bf16 rows: 64B/row, chunks at 16B alignment (SADDR + 32-bit voffset)
            const unsigned off = ((unsigned)gidx[k] << 6) + (unsigned)(half * 32);
            v8bf h0 = *(const v8bf*)(fbase + off);
            v8bf h1 = *(const v8bf*)(fbase + off + 16);
            a = __builtin_shufflevector(h0, h1, 0, 1, 2, 3, 4, 5, 6, 7,
                                                8, 9, 10, 11, 12, 13, 14, 15);
        } else {
            const unsigned off = ((unsigned)gidx[k] << 7) + (unsigned)(half * 64);
            f32x8 fa0 = *(const f32x8*)(fbase + off);
            f32x8 fa1 = *(const f32x8*)(fbase + off + 32);
            v8bf h0 = __builtin_convertvector(fa0, v8bf);
            v8bf h1 = __builtin_convertvector(fa1, v8bf);
            a = __builtin_shufflevector(h0, h1, 0, 1, 2, 3, 4, 5, 6, 7,
                                                8, 9, 10, 11, 12, 13, 14, 15);
        }

        // B fragments (32x16): lane holds column n=lm, cin range half*16..+15
        // 32B-aligned contiguous pair-words -> 2x ds_load_b128 each
        const unsigned int* wbk = &ldsW[(k * COUT) * (CIN / 2)];
        v16bf b0 = __builtin_bit_cast(v16bf,
                     *(const u32x8*)(wbk + lm        * (CIN / 2) + half * 8));
        v16bf b1 = __builtin_bit_cast(v16bf,
                     *(const u32x8*)(wbk + (16 + lm) * (CIN / 2) + half * 8));

        acc0 = __builtin_amdgcn_wmma_f32_16x16x32_bf16(
                   false, a, false, b0, (short)0, acc0, false, false);
        acc1 = __builtin_amdgcn_wmma_f32_16x16x32_bf16(
                   false, a, false, b1, (short)0, acc1, false, false);
    }

    // C/D layout: VGPR r holds M = r + 8*half, lane%16 is N
#pragma unroll
    for (int r = 0; r < 8; ++r) {
        const int m = r + half * 8;
        const int row = base + m;
        if (row < n) {
            float* op = xout + (long long)row * COUT;
            op[lm]      = acc0[r];
            op[16 + lm] = acc1[r];
        }
    }
}

// ---------------------------------------------------------------------------
// 2) per-(batch, channel) sum / sumsq / count via LDS float atomics
// ---------------------------------------------------------------------------
#define VOX_PER_BLOCK 2048
__global__ __launch_bounds__(256) void stats_kernel(
    const float* __restrict__ x, const int* __restrict__ bidx,
    float* gsum, float* gsqs, float* gcnt, int n)
{
    __shared__ float ssum[NB * COUT];
    __shared__ float ssqs[NB * COUT];
    __shared__ float scnt[NB];
    const int tid = threadIdx.x;
    for (int i = tid; i < NB * COUT; i += 256) { ssum[i] = 0.f; ssqs[i] = 0.f; }
    if (tid < NB) scnt[tid] = 0.f;
    __syncthreads();

    const int v0 = blockIdx.x * VOX_PER_BLOCK + tid;
#pragma unroll
    for (int it = 0; it < VOX_PER_BLOCK / 256; ++it) {
        const int v = v0 + it * 256;
        if (v < n) {
            const int b = bidx[v];
            atomicAdd(&scnt[b], 1.0f);
            const float4* xp = (const float4*)(x + (long long)v * COUT);
#pragma unroll
            for (int q = 0; q < 8; ++q) {
                float4 f = xp[q];
                const int cb = b * COUT + q * 4;
                atomicAdd(&ssum[cb + 0], f.x); atomicAdd(&ssqs[cb + 0], f.x * f.x);
                atomicAdd(&ssum[cb + 1], f.y); atomicAdd(&ssqs[cb + 1], f.y * f.y);
                atomicAdd(&ssum[cb + 2], f.z); atomicAdd(&ssqs[cb + 2], f.z * f.z);
                atomicAdd(&ssum[cb + 3], f.w); atomicAdd(&ssqs[cb + 3], f.w * f.w);
            }
        }
    }
    __syncthreads();
    for (int i = tid; i < NB * COUT; i += 256) {
        atomicAdd(&gsum[i], ssum[i]);
        atomicAdd(&gsqs[i], ssqs[i]);
    }
    if (tid < NB) atomicAdd(&gcnt[tid], scnt[tid]);
}

// ---------------------------------------------------------------------------
// 3) styles + fold instance-norm into per-(b,c) scale/shift   (1 block)
// ---------------------------------------------------------------------------
__global__ __launch_bounds__(256) void finalize_kernel(
    const float* __restrict__ w,     // [8, 512]
    const float* __restrict__ affW,  // [32, 512]
    const float* __restrict__ affB,  // [32]
    const float* __restrict__ iw,    // [32]
    const float* __restrict__ ib,    // [32]
    const float* gsum, const float* gsqs, const float* gcnt,
    float* scale, float* shift)
{
    const int t = threadIdx.x;          // 0..255 == (b, c)
    const int b = t >> 5, c = t & 31;
    const float gain = 0.04419417382415922f;   // 1/sqrt(512)

    const float* wr = w    + b * WDIM;
    const float* ar = affW + c * WDIM;
    float acc = 0.f;
    for (int j = 0; j < WDIM; ++j) acc += wr[j] * ar[j];
    const float style = acc * gain + affB[c];

    const float cnt  = gcnt[b];
    const float mean = gsum[t] / cnt;
    const float var  = gsqs[t] / cnt - mean * mean;
    const float inv  = rsqrtf(var + EPSV);
    const float g    = inv * iw[c];
    scale[t] = g * style;
    shift[t] = (ib[c] - mean * g) * style;
}

// ---------------------------------------------------------------------------
// 4) in-place modulate + leaky-ReLU
// ---------------------------------------------------------------------------
__global__ __launch_bounds__(256) void apply_kernel(
    float* __restrict__ x, const int* __restrict__ bidx,
    const float* __restrict__ scale, const float* __restrict__ shift, int n)
{
    __shared__ float ssc[NB * COUT];
    __shared__ float ssh[NB * COUT];
    const int tid = threadIdx.x;
    ssc[tid] = scale[tid];
    ssh[tid] = shift[tid];
    __syncthreads();

    const int v = blockIdx.x * 256 + tid;
    if (v >= n) return;
    const int b = bidx[v];
    float4* xp = (float4*)(x + (long long)v * COUT);
#pragma unroll
    for (int q = 0; q < 8; ++q) {
        float4 f = xp[q];
        const int cb = b * COUT + q * 4;
        float y0 = f.x * ssc[cb + 0] + ssh[cb + 0];
        float y1 = f.y * ssc[cb + 1] + ssh[cb + 1];
        float y2 = f.z * ssc[cb + 2] + ssh[cb + 2];
        float y3 = f.w * ssc[cb + 3] + ssh[cb + 3];
        f.x = (y0 >= 0.f) ? y0 : SLOPE * y0;
        f.y = (y1 >= 0.f) ? y1 : SLOPE * y1;
        f.z = (y2 >= 0.f) ? y2 : SLOPE * y2;
        f.w = (y3 >= 0.f) ? y3 : SLOPE * y3;
        xp[q] = f;
    }
}

// ---------------------------------------------------------------------------
extern "C" void kernel_launch(void* const* d_in, const int* in_sizes, int n_in,
                              void* d_out, int out_size, void* d_ws, size_t ws_size,
                              hipStream_t stream) {
    const float* feats = (const float*)d_in[0];
    const float* w     = (const float*)d_in[1];
    const float* convW = (const float*)d_in[2];
    const float* convB = (const float*)d_in[3];
    const float* affW  = (const float*)d_in[4];
    const float* affB  = (const float*)d_in[5];
    const float* iw    = (const float*)d_in[6];
    const float* ib    = (const float*)d_in[7];
    const int*   nidx  = (const int*)d_in[8];
    const int*   bidx  = (const int*)d_in[9];

    float* x = (float*)d_out;                 // conv output lives in d_out, finished in place

    const int n      = in_sizes[0] / CIN;     // number of voxels
    const int nTiles = (n + 15) / 16;

    // workspace layout: [stats: 1032 floats][pad to 8KB][bf16 feats: N*32]
    float* wsf   = (float*)d_ws;
    float* gsum  = wsf;
    float* gsqs  = wsf + 256;
    float* gcnt  = wsf + 512;
    float* scl   = wsf + 520;
    float* shf   = wsf + 776;
    const int nzero = 1032;

    bf16* fb = (bf16*)((char*)d_ws + 8192);
    const size_t need = 8192 + (size_t)n * CIN * sizeof(bf16);
    const bool useBf16Feats = (ws_size >= need);

    zero_ws_kernel<<<(nzero + 255) / 256, 256, 0, stream>>>(wsf, nzero);

    const int convGrid = (nTiles + TILES_PER_BLOCK - 1) / TILES_PER_BLOCK;
    if (useBf16Feats) {
        const int total8 = n * CIN / 8;
        cvt_feats_kernel<<<(total8 + 255) / 256, 256, 0, stream>>>(feats, fb, total8);
        conv_wmma_kernel<true><<<convGrid, 256, 0, stream>>>(
            (const void*)fb, convW, convB, nidx, x, n, nTiles);
    } else {
        conv_wmma_kernel<false><<<convGrid, 256, 0, stream>>>(
            (const void*)feats, convW, convB, nidx, x, n, nTiles);
    }

    stats_kernel<<<(n + VOX_PER_BLOCK - 1) / VOX_PER_BLOCK, 256, 0, stream>>>(
        x, bidx, gsum, gsqs, gcnt, n);

    finalize_kernel<<<1, 256, 0, stream>>>(w, affW, affB, iw, ib,
                                           gsum, gsqs, gcnt, scl, shf);

    apply_kernel<<<(n + 255) / 256, 256, 0, stream>>>(x, bidx, scl, shf, n);
}
